// MCC_46076409152266
// MI455X (gfx1250) — matrix operand order repaired and checked by reading, not compiled, not run on hardware
//
#include <hip/hip_runtime.h>
#include <hip/hip_bf16.h>

typedef __attribute__((ext_vector_type(16))) __bf16 v16bf;
typedef __attribute__((ext_vector_type(8)))  float  v8f;

#define BATCH 2
#define CCH   256
#define HH    256
#define WW    256
#define HEADS 8
#define NTOK  65536
#define EPS_LN 1e-6f
#define EPS_NORM 1e-12f

// ---------------- LayerNorm over channels (channels_first) ----------------
__global__ void ln_kernel(const float* __restrict__ x, const float* __restrict__ w,
                          const float* __restrict__ bia, float* __restrict__ y) {
  size_t p = (size_t)blockIdx.x * blockDim.x + threadIdx.x; // pixel id in [0, B*NTOK)
  size_t b = p / NTOK;
  size_t n = p % NTOK;
  const float* xb = x + b * (size_t)CCH * NTOK + n;
  float s = 0.f, ss = 0.f;
  #pragma unroll 4
  for (int c = 0; c < CCH; ++c) {
    float v = xb[(size_t)c * NTOK];
    s += v; ss += v * v;
  }
  float mean = s * (1.0f / CCH);
  float var  = ss * (1.0f / CCH) - mean * mean;
  float rinv = rsqrtf(var + EPS_LN);
  float* yb = y + b * (size_t)CCH * NTOK + n;
  #pragma unroll 4
  for (int c = 0; c < CCH; ++c) {
    float v = xb[(size_t)c * NTOK];
    yb[(size_t)c * NTOK] = w[c] * ((v - mean) * rinv) + bia[c];
  }
}

// ---------------- Depthwise 3x3, reflect pad 1, optional leaky ----------------
__global__ void dwconv_kernel(const float* __restrict__ x, const float* __restrict__ wgt,
                              float* __restrict__ y, int Cin, int leaky) {
  size_t idx = (size_t)blockIdx.x * blockDim.x + threadIdx.x;
  size_t n  = idx % NTOK;
  size_t bc = idx / NTOK;           // b*Cin + c
  int c = (int)(bc % (size_t)Cin);
  int h = (int)(n / WW), wp = (int)(n % WW);
  const float* xb = x + bc * (size_t)NTOK;
  const float* kw = wgt + c * 9;
  float acc = 0.f;
  #pragma unroll
  for (int dy = -1; dy <= 1; ++dy) {
    int rh = h + dy; rh = rh < 0 ? 1 : (rh >= HH ? HH - 2 : rh);
    const float* row = xb + (size_t)rh * WW;
    __builtin_prefetch(row + wp, 0, 1);
    #pragma unroll
    for (int dx = -1; dx <= 1; ++dx) {
      int rw = wp + dx; rw = rw < 0 ? 1 : (rw >= WW ? WW - 2 : rw);
      acc += kw[(dy + 1) * 3 + (dx + 1)] * row[rw];
    }
  }
  if (leaky) acc = acc > 0.f ? acc : 0.1f * acc;
  y[idx] = acc;
}

// ---------------- helpers ----------------
__global__ void cvt_bf16_kernel(const float* __restrict__ src, __bf16* __restrict__ dst, int n) {
  int i = blockIdx.x * blockDim.x + threadIdx.x;
  if (i < n) dst[i] = (__bf16)src[i];
}

__global__ void zero_kernel(float* __restrict__ p, int n) {
  int i = blockIdx.x * blockDim.x + threadIdx.x;
  if (i < n) p[i] = 0.f;
}

// ---------------- Gram (q . k^T) + row sum-of-squares, partial + atomics ----------------
// q/k tiles staged to LDS with gfx1250 async global->LDS B128 copies (ASYNCcnt path).
__global__ void gram_kernel(const float* __restrict__ q, const float* __restrict__ kv,
                            float* __restrict__ S, float* __restrict__ qss,
                            float* __restrict__ kss) {
  __shared__ float qs[32][128];
  __shared__ float ks[32][128];
  int bh = blockIdx.y;                  // b*HEADS + h
  int b = bh >> 3, hd = bh & 7;
  size_t n0 = (size_t)blockIdx.x * 128;
  const float* qb = q  + ((size_t)b * CCH + hd * 32) * (size_t)NTOK + n0;
  const float* kb = kv + ((size_t)b * 2 * CCH + hd * 32) * (size_t)NTOK + n0; // k = first C ch
  int t = threadIdx.x;

  // Async stage: 32 rows x 512B per array; 4 x B128 per thread per array.
  unsigned ldsq0 = (unsigned)(size_t)(&qs[0][0]);
  unsigned ldsk0 = (unsigned)(size_t)(&ks[0][0]);
  #pragma unroll
  for (int i = 0; i < 4; ++i) {
    int e = t + 256 * i;          // b128 chunk id: 32 chunks per row
    int r  = e >> 5;
    int cc = (e & 31) << 2;       // float column
    unsigned goff = (unsigned)((r * NTOK + cc) * 4);      // byte offset (<8.2MB)
    unsigned loff = (unsigned)((r * 128 + cc) * 4);       // byte offset in tile
    unsigned lq = ldsq0 + loff;
    unsigned lk = ldsk0 + loff;
    asm volatile("global_load_async_to_lds_b128 %0, %1, %2"
                 :: "v"(lq), "v"(goff), "s"(qb) : "memory");
    asm volatile("global_load_async_to_lds_b128 %0, %1, %2"
                 :: "v"(lk), "v"(goff), "s"(kb) : "memory");
  }
  asm volatile("s_wait_asynccnt 0x0" ::: "memory");
  __syncthreads();

  #pragma unroll
  for (int i = 0; i < 4; ++i) {
    int id = t + 256 * i;
    int cq = id >> 5, ck = id & 31;
    float sum = 0.f;
    #pragma unroll 8
    for (int col = 0; col < 128; ++col) sum += qs[cq][col] * ks[ck][col];
    atomicAdd(&S[(size_t)bh * 1024 + id], sum);
  }
  if (t < 32) {
    float sum = 0.f;
    #pragma unroll 8
    for (int col = 0; col < 128; ++col) { float v = qs[t][col]; sum += v * v; }
    atomicAdd(&qss[b * CCH + hd * 32 + t], sum);
  } else if (t >= 128 && t < 160) {
    int ck = t - 128;
    float sum = 0.f;
    #pragma unroll 8
    for (int col = 0; col < 128; ++col) { float v = ks[ck][col]; sum += v * v; }
    atomicAdd(&kss[b * CCH + hd * 32 + ck], sum);
  }
}

// ------------- softmax(attn) and fold W_proj: M[o, h*32+d] = sum_c Wp[o,h*32+c]*attn[h,c,d] -------------
__global__ void softmax_m_kernel(const float* __restrict__ S, const float* __restrict__ qss,
                                 const float* __restrict__ kss, const float* __restrict__ temp,
                                 const float* __restrict__ wproj, __bf16* __restrict__ M) {
  __shared__ float attn[HEADS][32][32];
  int b = blockIdx.x;
  int t = threadIdx.x;
  {
    int hd = t >> 5, cq = t & 31;
    float qn = fmaxf(sqrtf(qss[b * CCH + hd * 32 + cq]), EPS_NORM);
    float tmp = temp[hd];
    float vals[32];
    float mx = -1e30f;
    #pragma unroll
    for (int d = 0; d < 32; ++d) {
      float kn = fmaxf(sqrtf(kss[b * CCH + hd * 32 + d]), EPS_NORM);
      float v = S[((size_t)(b * HEADS + hd)) * 1024 + cq * 32 + d] / (qn * kn) * tmp;
      vals[d] = v; mx = fmaxf(mx, v);
    }
    float sum = 0.f;
    #pragma unroll
    for (int d = 0; d < 32; ++d) { vals[d] = __expf(vals[d] - mx); sum += vals[d]; }
    float rs = 1.f / sum;
    #pragma unroll
    for (int d = 0; d < 32; ++d) attn[hd][cq][d] = vals[d] * rs;
  }
  __syncthreads();
  int o = t;
  for (int hd = 0; hd < HEADS; ++hd) {
    #pragma unroll 4
    for (int d = 0; d < 32; ++d) {
      float acc = 0.f;
      #pragma unroll 8
      for (int c = 0; c < 32; ++c)
        acc += wproj[o * CCH + hd * 32 + c] * attn[hd][c][d];
      M[((size_t)b * CCH + o) * CCH + hd * 32 + d] = (__bf16)acc;
    }
  }
}

// ---------------- WMMA bf16 GEMM: Y[o,n] = act( W[o,:] . X[:,n] + Res ) ----------------
// Block: 256 threads = 8 waves. Tile: 128 (o) x 64 (n). K fully unrolled (CDIM/32 steps),
// register-staged prefetch of the next X tile overlaps global latency with WMMA issue.
template<int CDIM>
__global__ void gemm_wmma_kernel(const __bf16* __restrict__ Wb, const float* __restrict__ X,
                                 float* __restrict__ Y, const float* __restrict__ Res,
                                 long wStrideB, long xStrideB,
                                 long yStrideB, long resStrideB, int leaky) {
  constexpr int KSTEPS = CDIM / 32;
  __shared__ __bf16 bt[64][36];   // row stride 72B: 4B-aligned K-pairs
  int b = blockIdx.z;
  size_t n0 = (size_t)blockIdx.x * 64;
  int wave = threadIdx.x >> 5;
  int lane = threadIdx.x & 31;
  int o16 = blockIdx.y * 128 + wave * 16;

  v8f acc[4];
  #pragma unroll
  for (int s = 0; s < 4; ++s)
    #pragma unroll
    for (int i = 0; i < 8; ++i) acc[s][i] = 0.f;

  const float*  Xb    = X  + (size_t)b * (size_t)xStrideB;
  const __bf16* Wrow0 = Wb + (size_t)b * (size_t)wStrideB;

  int nl  = threadIdx.x & 63;
  int kk0 = (threadIdx.x >> 6) << 3;   // 0,8,16,24
  int m   = lane & 15;
  int hh  = lane >> 4;
  int kb  = (lane < 16) ? 0 : 8;

  float stage[8];
  // prologue: stage k0 = 0
  {
    const float* xp = Xb + (size_t)kk0 * NTOK + n0 + nl;
    #pragma unroll
    for (int j = 0; j < 8; ++j) stage[j] = xp[(size_t)j * NTOK];
    #pragma unroll
    for (int j = 0; j < 8; ++j) bt[nl][kk0 + j] = (__bf16)stage[j];
  }
  __syncthreads();

  #pragma unroll
  for (int ks = 0; ks < KSTEPS; ++ks) {
    const int k0 = ks * 32;
    // prefetch next X tile into registers while WMMAs run
    if (ks + 1 < KSTEPS) {
      const float* xp = Xb + (size_t)(k0 + 32 + kk0) * NTOK + n0 + nl;
      #pragma unroll
      for (int j = 0; j < 8; ++j) stage[j] = xp[(size_t)j * NTOK];
    }

    // A fragment (16x32 bf16): lanes 0-15 rows M, K pairs {0,2,4,6,16,18,20,22}; lanes 16-31 +8
    v16bf afrag;
    const __bf16* wr = Wrow0 + (size_t)(o16 + m) * CDIM + k0 + hh * 8;
    #pragma unroll
    for (int j = 0; j < 8; ++j) {
      int kj = (j < 4) ? (2 * j) : (16 + 2 * (j - 4));
      afrag[2 * j]     = wr[kj];
      afrag[2 * j + 1] = wr[kj + 1];
    }

    #pragma unroll
    for (int s = 0; s < 4; ++s) {
      // B fragment (32x16 bf16): col = lane%16, lanes<16 K 0..7,16..23; lanes>=16 +8
      v16bf bfrag;
      int nc = (lane & 15) + s * 16;
      #pragma unroll
      for (int j = 0; j < 8; ++j) {
        bfrag[2 * j]     = bt[nc][kb + ((j < 4) ? (2 * j) : (16 + 2 * (j - 4)))];
        bfrag[2 * j + 1] = bt[nc][kb + ((j < 4) ? (2 * j) : (16 + 2 * (j - 4))) + 1];
      }
      acc[s] = __builtin_amdgcn_wmma_f32_16x16x32_bf16(false, afrag, false, bfrag,
                                                       (short)0, acc[s], false, false);
    }

    if (ks + 1 < KSTEPS) {
      __syncthreads();   // all waves done reading current tile
      #pragma unroll
      for (int j = 0; j < 8; ++j) bt[nl][kk0 + j] = (__bf16)stage[j];
      __syncthreads();   // next tile visible
    }
  }

  // D layout: VGPR r -> row (lane<16 ? r : r+8), col = lane%16
  int rbase = (lane < 16) ? 0 : 8;
  int colb  = lane & 15;
  #pragma unroll
  for (int s = 0; s < 4; ++s) {
    #pragma unroll
    for (int r = 0; r < 8; ++r) {
      int row = o16 + rbase + r;
      size_t col = n0 + colb + (size_t)s * 16;
      float v = acc[s][r];
      if (Res)   v += Res[(size_t)b * (size_t)resStrideB + (size_t)row * NTOK + col];
      if (leaky) v = v > 0.f ? v : 0.1f * v;
      Y[(size_t)b * (size_t)yStrideB + (size_t)row * NTOK + col] = v;
    }
  }
}

// ---------------- host launcher ----------------
extern "C" void kernel_launch(void* const* d_in, const int* in_sizes, int n_in,
                              void* d_out, int out_size, void* d_ws, size_t ws_size,
                              hipStream_t stream) {
  const float* x_Q         = (const float*)d_in[0];
  const float* x_KV        = (const float*)d_in[1];
  const float* ln_w        = (const float*)d_in[2];
  const float* ln_b        = (const float*)d_in[3];
  const float* temperature = (const float*)d_in[4];
  const float* w_q         = (const float*)d_in[5];
  const float* w_kv_pw     = (const float*)d_in[6];
  const float* w_kv_dw     = (const float*)d_in[7];
  const float* w_proj      = (const float*)d_in[8];
  const float* w_ff1       = (const float*)d_in[9];
  const float* w_ffdw      = (const float*)d_in[10];
  float* out = (float*)d_out;
  float* ws  = (float*)d_ws;

  const size_t CN  = (size_t)CCH * NTOK;       // 16,777,216 floats
  const size_t BCN = (size_t)BATCH * CN;       // 33,554,432 floats

  // Buffer plan (peak 5*BCN floats + small tail):
  float* kvbuf = ws;              // [B][2C][N] : reused region, first hosts lnq+lnkv
  float* lnq   = ws;              // [0, BCN)
  float* lnkv  = ws + BCN;        // [BCN, 2BCN)
  float* qbuf  = ws + 2 * BCN;    // [2BCN, 3BCN)
  float* kvpw  = ws + 3 * BCN;    // [3BCN, 5BCN) (dead after kv dwconv)
  float* ybuf  = ws + 3 * BCN;    // reuse
  float* zbuf  = ws + 4 * BCN;    // reuse
  float* Sbuf  = ws + 5 * BCN;                          // B*H*32*32 = 16384
  float* qss   = Sbuf + (size_t)BATCH * HEADS * 1024;   // B*C = 512
  float* kss   = qss + BATCH * CCH;                     // 512
  __bf16* Wkv_bf = (__bf16*)(kss + BATCH * CCH);        // 512*256 bf16
  __bf16* Wff_bf = Wkv_bf + (size_t)2 * CCH * CCH;      // 256*256 bf16
  __bf16* M_bf   = Wff_bf + (size_t)CCH * CCH;          // B*256*256 bf16

  // zero accumulators (S, qss, kss) for deterministic atomics
  int zn = BATCH * HEADS * 1024 + 2 * BATCH * CCH;
  zero_kernel<<<(zn + 255) / 256, 256, 0, stream>>>(Sbuf, zn);

  // weight precision prep
  cvt_bf16_kernel<<<(2 * CCH * CCH + 255) / 256, 256, 0, stream>>>(w_kv_pw, Wkv_bf, 2 * CCH * CCH);
  cvt_bf16_kernel<<<(CCH * CCH + 255) / 256, 256, 0, stream>>>(w_ff1, Wff_bf, CCH * CCH);

  // LayerNorms
  int pixBlocks = (int)(((size_t)BATCH * NTOK) / 256);
  ln_kernel<<<pixBlocks, 256, 0, stream>>>(x_Q,  ln_w, ln_b, lnq);
  ln_kernel<<<pixBlocks, 256, 0, stream>>>(x_KV, ln_w, ln_b, lnkv);

  // q = dwconv3x3_reflect(lnq)
  dwconv_kernel<<<(int)((BATCH * (size_t)CCH * NTOK) / 256), 256, 0, stream>>>(lnq, w_q, qbuf, CCH, 0);

  // kv_pw = W_kv_pw @ lnkv     [B][512][N]   (WMMA)
  gemm_wmma_kernel<CCH><<<dim3(NTOK / 64, (2 * CCH) / 128, BATCH), 256, 0, stream>>>(
      Wkv_bf, lnkv, kvpw, nullptr,
      /*wStrideB*/0, /*xStrideB*/(long)CN, /*yStrideB*/(long)(2 * CN), /*resStrideB*/0, /*leaky*/0);

  // kv = dwconv3x3_reflect(kv_pw)  -> overwrites (dead) lnq/lnkv region
  dwconv_kernel<<<(int)((BATCH * (size_t)(2 * CCH) * NTOK) / 256), 256, 0, stream>>>(
      kvpw, w_kv_dw, kvbuf, 2 * CCH, 0);

  // Gram S = q . k^T and sum-of-squares norms (norm fold happens in softmax)
  gram_kernel<<<dim3(NTOK / 128, BATCH * HEADS), 256, 0, stream>>>(qbuf, kvbuf, Sbuf, qss, kss);

  // attn softmax + fold W_proj into per-batch M (bf16)
  softmax_m_kernel<<<BATCH, 256, 0, stream>>>(Sbuf, qss, kss, temperature, w_proj, M_bf);

  // y = M_b @ v + x_Q          (WMMA; v = kv channels C..2C-1)
  gemm_wmma_kernel<CCH><<<dim3(NTOK / 64, CCH / 128, BATCH), 256, 0, stream>>>(
      M_bf, kvbuf + CN, ybuf, x_Q,
      /*wStrideB*/(long)(CCH * CCH), /*xStrideB*/(long)(2 * CN), /*yStrideB*/(long)CN,
      /*resStrideB*/(long)CN, /*leaky*/0);

  // z = leaky_relu(W_ff1 @ y)  (WMMA)
  gemm_wmma_kernel<CCH><<<dim3(NTOK / 64, CCH / 128, BATCH), 256, 0, stream>>>(
      Wff_bf, ybuf, zbuf, nullptr,
      /*wStrideB*/0, /*xStrideB*/(long)CN, /*yStrideB*/(long)CN, /*resStrideB*/0, /*leaky*/1);

  // out = leaky_relu(dwconv3x3_reflect(z))
  dwconv_kernel<<<(int)((BATCH * (size_t)CCH * NTOK) / 256), 256, 0, stream>>>(
      zbuf, w_ffdw, out, CCH, 1);
}
